// DenoisingNetwork_25202868093635
// MI455X (gfx1250) — compile-verified, matrix operand order checked by reading, not run on hardware
//
#include <hip/hip_runtime.h>

#define SDIM 256
#define VDIM 64
#define RBF_DIM 32
#define N_NODES 4096
#define N_GRAPHS 128
#define E_LOCAL 49152
#define E_GLOBAL 126976
#define E_TOT (E_LOCAL + E_GLOBAL)
#define NUM_LAYERS 5
#define CUTOFF_F 7.5f
#define PI_F 3.14159265358979f

typedef __attribute__((ext_vector_type(16))) __bf16 v16bf;
typedef __attribute__((ext_vector_type(8)))  float  v8f;
typedef __attribute__((ext_vector_type(4)))  unsigned int v4u;
typedef __attribute__((ext_vector_type(8)))  int v8i;
typedef __attribute__((ext_vector_type(4)))  int v4i;

__device__ __forceinline__ float siluf(float x) { return x / (1.0f + __expf(-x)); }

// Load a 16x32 bf16 fragment (A layout: lane&15 = row, K split across VGPR halves).
// Works for B as well when the weight matrix is stored transposed [N][K].
__device__ __forceinline__ v16bf load_frag(const __bf16* base, int ldk, int lane) {
  int half = (lane >> 4) & 1;
  int r = lane & 15;
  const __bf16* p = base + (size_t)r * ldk + half * 8;
  v16bf f;
#pragma unroll
  for (int i = 0; i < 8; ++i) f[i] = p[i];          // K = half*8 + 0..7
#pragma unroll
  for (int i = 0; i < 8; ++i) f[8 + i] = p[16 + i]; // K = 16 + half*8 + 0..7
  return f;
}

__device__ __forceinline__ v8f wmma_bf16(v16bf a, v16bf b, v8f c) {
  return __builtin_amdgcn_wmma_f32_16x16x32_bf16(false, a, false, b, (short)0, c,
                                                 false, false);
}

// TDM gather: pull 16 rows (16-bit indices) of 256 x bf16 each from a
// [4096 x 256] bf16 tensor in global memory into contiguous LDS.
// D# layout per CDNA5 ISA ch.8 (gather mode, data_size=2B).
// This toolchain exposes the 6-arg builtin form (extra int32x8 operand).
__device__ __forceinline__ void tdm_gather16(unsigned lds_addr, const __bf16* gbase,
                                             const int* idx /* 16 row indices */) {
  unsigned long long ga = (unsigned long long)(size_t)gbase;
  v4u g0;
  g0[0] = 1u | (1u << 31);                        // count=1, gather_mode=1, 16b idx
  g0[1] = lds_addr;                               // lds_addr [63:32]
  g0[2] = (unsigned)(ga & 0xffffffffu);           // global_addr lo
  g0[3] = (unsigned)((ga >> 32) & 0x01ffffffu) | (2u << 30);  // addr hi, type=2
  v8i g1;
  g1[0] = (1 << 16);                              // wg_mask=0, data_size=1 (2 bytes)
  g1[1] = (int)((256u & 0xffffu) << 16);          // tensor_dim0[15:0] = 256
  g1[2] = (int)((4096u & 0xffffu) << 16);         // dim0 hi=0 | tensor_dim1[15:0]=4096
  g1[3] = (int)(256u << 16);                      // dim1 hi=0 | tile_dim0=256
  g1[4] = 16;                                     // tile_dim1 = 16 valid indices
  g1[5] = 256;                                    // tensor_dim0_stride lo
  g1[6] = 0;                                      // stride0 hi | stride1 lo (ignored)
  g1[7] = 0;
  v4i g2, g3;
#pragma unroll
  for (int j = 0; j < 4; ++j)
    g2[j] = (idx[2 * j] & 0xffff) | (idx[2 * j + 1] << 16);
#pragma unroll
  for (int j = 0; j < 4; ++j)
    g3[j] = (idx[8 + 2 * j] & 0xffff) | (idx[8 + 2 * j + 1] << 16);
  v8i g4 = {};                                    // extra operand (zero)
  __builtin_amdgcn_tensor_load_to_lds(g0, g1, g2, g3, g4, 0);
}

// ---------------------------------------------------------------- weights prep
__global__ void transpose_to_bf16(const float* __restrict__ W, __bf16* __restrict__ Wt,
                                  int K, int N, int Kp) {
  int idx = blockIdx.x * 256 + threadIdx.x;
  if (idx >= N * Kp) return;
  int n = idx / Kp, k = idx % Kp;
  float v = (k < K) ? W[(size_t)k * N + n] : 0.0f;
  Wt[(size_t)n * Kp + k] = (__bf16)v;
}

// ---------------------------------------------------------------- pos centering
__global__ void accum_graph3(const float* __restrict__ p3, const int* __restrict__ batch,
                             float* __restrict__ gsum) {
  int n = blockIdx.x * 256 + threadIdx.x;
  if (n >= N_NODES) return;
  int g = batch[n];
  atomicAdd(&gsum[g * 4 + 0], p3[n * 3 + 0]);
  atomicAdd(&gsum[g * 4 + 1], p3[n * 3 + 1]);
  atomicAdd(&gsum[g * 4 + 2], p3[n * 3 + 2]);
  atomicAdd(&gsum[g * 4 + 3], 1.0f);
}

__global__ void center_pos(const float* __restrict__ pos, const int* __restrict__ batch,
                           const float* __restrict__ gsum, float* __restrict__ pos_c) {
  int n = blockIdx.x * 256 + threadIdx.x;
  if (n >= N_NODES) return;
  int g = batch[n];
  float c = fmaxf(gsum[g * 4 + 3], 1.0f);
#pragma unroll
  for (int d = 0; d < 3; ++d)
    pos_c[n * 3 + d] = pos[n * 3 + d] - gsum[g * 4 + d] / c;
}

// ---------------------------------------------------------------- edge geometry
__global__ void edge_geom(const int* __restrict__ eil, const int* __restrict__ eig,
                          const float* __restrict__ pos_c,
                          int* __restrict__ src_all, int* __restrict__ dst_all,
                          float* __restrict__ env_all, float* __restrict__ rn_all,
                          __bf16* __restrict__ rbf_bf, float* __restrict__ deg) {
  int e = blockIdx.x * 256 + threadIdx.x;
  if (e >= E_TOT) return;
  int s, t; bool local = (e < E_LOCAL);
  if (local) { s = eil[e]; t = eil[E_LOCAL + e]; }
  else       { int e2 = e - E_LOCAL; s = eig[e2]; t = eig[E_GLOBAL + e2]; }
  src_all[e] = s; dst_all[e] = t;
  float rx = pos_c[t * 3 + 0] - pos_c[s * 3 + 0];
  float ry = pos_c[t * 3 + 1] - pos_c[s * 3 + 1];
  float rz = pos_c[t * 3 + 2] - pos_c[s * 3 + 2];
  float d2 = fmaxf(rx * rx + ry * ry + rz * rz, 1e-6f);
  float d = sqrtf(d2);
  float inv = 1.0f / d;
  rn_all[(size_t)e * 3 + 0] = rx * inv;
  rn_all[(size_t)e * 3 + 1] = ry * inv;
  rn_all[(size_t)e * 3 + 2] = rz * inv;
  float env;
  if (local) env = (d < CUTOFF_F) ? 0.5f * (__cosf(PI_F * d / CUTOFF_F) + 1.0f) : 0.0f;
  else       env = 1.0f;
  env_all[e] = env;
  float scale = (float)RBF_DIM / CUTOFF_F;
#pragma unroll
  for (int i = 0; i < RBF_DIM; ++i) {
    float ce = CUTOFF_F * (float)i / (float)(RBF_DIM - 1);
    float dd = d - ce;
    rbf_bf[(size_t)e * RBF_DIM + i] = (__bf16)__expf(-scale * dd * dd);
  }
  atomicAdd(&deg[t], 1.0f);
}

// ---------------------------------------------------------------- embedding (K=16)
__global__ void embed1(const float* __restrict__ x, const float* __restrict__ t,
                       const int* __restrict__ batch,
                       const float* __restrict__ timeW, const float* __restrict__ timeb,
                       const float* __restrict__ atomW, const float* __restrict__ atomb,
                       __bf16* __restrict__ tmp_bf) {
  int idx = blockIdx.x * 256 + threadIdx.x;
  if (idx >= N_NODES * SDIM) return;
  int n = idx >> 8, c = idx & 255;
  int g = batch[n];
  float acc = atomb[c] + t[g] * timeW[c] + timeb[c];
#pragma unroll
  for (int k = 0; k < 16; ++k) acc += x[n * 16 + k] * atomW[k * SDIM + c];
  tmp_bf[idx] = (__bf16)acc;
}

// ---------------------------------------------------------------- generic node GEMM
__global__ void gemm_nodes(const __bf16* __restrict__ A, const __bf16* __restrict__ Wt,
                           const float* __restrict__ bias, float* __restrict__ outf,
                           __bf16* __restrict__ outb, int K, int N, int do_silu) {
  int m0 = blockIdx.x * 16;
  int tid = threadIdx.x, lane = tid & 31, wave = tid >> 5;
  int half = (lane >> 4) & 1;
  for (int nt = wave; nt < (N >> 4); nt += 8) {
    v8f acc = {};
    for (int kc = 0; kc < K; kc += 32) {
      v16bf a = load_frag(A + (size_t)m0 * K + kc, K, lane);
      v16bf b = load_frag(Wt + (size_t)nt * 16 * K + kc, K, lane);
      acc = wmma_bf16(a, b, acc);
    }
    int n = nt * 16 + (lane & 15);
    float bb = bias ? bias[n] : 0.0f;
#pragma unroll
    for (int g = 0; g < 8; ++g) {
      int row = m0 + g + 8 * half;
      float v = acc[g] + bb;
      if (do_silu) v = siluf(v);
      size_t o = (size_t)row * N + n;
      outf[o] = v;
      if (outb) outb[o] = (__bf16)v;
    }
  }
}

// ---------------------------------------------------------------- fused edge kernel
__global__ void layer_edge(const int* __restrict__ src_all, const int* __restrict__ dst_all,
                           const float* __restrict__ rn_all, const float* __restrict__ env_all,
                           const __bf16* __restrict__ rbf_bf, const __bf16* __restrict__ s_bf,
                           const __bf16* __restrict__ v_bf,
                           const __bf16* __restrict__ WeT, const float* __restrict__ be,
                           const __bf16* __restrict__ WvT, const __bf16* __restrict__ WvvT,
                           float* __restrict__ agg_sum, float* __restrict__ v_sum) {
  __shared__ __bf16 featA[16 * 256];  // s[src]  (TDM gather target)
  __shared__ __bf16 featB[16 * 256];  // s[dst]  (TDM gather target)
  __shared__ __bf16 featR[16 * 32];   // rbf
  __shared__ __bf16 mb[16 * 256];     // message (bf16, for gate GEMM + scatter)
  __shared__ float  gatel[16 * 64];
  __shared__ __bf16 va[48 * 64];      // gathered v[src], rows = e*3+dim
  __shared__ float  hv[48 * 64];      // v[src] @ Wvv
  __shared__ float  rnl[48];
  __shared__ float  envl[16];
  __shared__ int    srcl[16], dstl[16];

  int e0 = blockIdx.x * 16;
  int tid = threadIdx.x, lane = tid & 31, wave = tid >> 5;
  int half = (lane >> 4) & 1;

  if (tid < 16) {
    int e = e0 + tid;
    srcl[tid] = src_all[e];
    dstl[tid] = dst_all[e];
    envl[tid] = env_all[e];
  }
  if (tid < 48) rnl[tid] = rn_all[(size_t)e0 * 3 + tid];

  // Wave 0 issues the two TDM row-gathers (s[src] rows -> featA, s[dst] -> featB).
  if (wave == 0) {
    int sidx[16], didx[16];
#pragma unroll
    for (int j = 0; j < 16; ++j) { sidx[j] = src_all[e0 + j]; didx[j] = dst_all[e0 + j]; }
    tdm_gather16((unsigned)(size_t)&featA[0], s_bf, sidx);
    tdm_gather16((unsigned)(size_t)&featB[0], s_bf, didx);
    __builtin_amdgcn_s_wait_tensorcnt(0);
  }
  __syncthreads();

  for (int i = tid; i < 16 * 32; i += 256) {
    int e = i >> 5, k = i & 31;
    featR[i] = rbf_bf[(size_t)(e0 + e) * RBF_DIM + k];
  }
  for (int i = tid; i < 48 * 64; i += 256) {
    int r = i >> 6, k = i & 63;
    int e = r / 3, dd = r - e * 3;
    va[i] = v_bf[((size_t)srcl[e] * 3 + dd) * VDIM + k];
  }
  __syncthreads();

  // GEMM1: m = silu([sA|sB|rbf] @ We + be) * env  (K = 256+256+32, N = 256)
  for (int nt = wave; nt < 16; nt += 8) {
    v8f acc = {};
    const __bf16* bbase = WeT + (size_t)nt * 16 * 544;
    __builtin_prefetch(bbase + 16 * 544, 0, 1);  // next tile's weights
    for (int kc = 0; kc < 256; kc += 32) {
      v16bf a = load_frag(featA + kc, 256, lane);
      v16bf b = load_frag(bbase + kc, 544, lane);
      acc = wmma_bf16(a, b, acc);
    }
    for (int kc = 0; kc < 256; kc += 32) {
      v16bf a = load_frag(featB + kc, 256, lane);
      v16bf b = load_frag(bbase + 256 + kc, 544, lane);
      acc = wmma_bf16(a, b, acc);
    }
    {
      v16bf a = load_frag(featR, 32, lane);
      v16bf b = load_frag(bbase + 512, 544, lane);
      acc = wmma_bf16(a, b, acc);
    }
    int n = nt * 16 + (lane & 15);
    float bb = be[n];
#pragma unroll
    for (int g = 0; g < 8; ++g) {
      int r = g + 8 * half;
      float v = siluf(acc[g] + bb) * envl[r];
      mb[r * 256 + n] = (__bf16)v;
    }
  }
  __syncthreads();

  // GEMM2 (waves 0-3): gate = m @ Wv (K=256, N=64)
  // GEMM3 (waves 4-7): hv = v[src] @ Wvv (M=48, K=64, N=64)
  if (wave < 4) {
    int nt = wave;
    v8f acc = {};
    for (int kc = 0; kc < 256; kc += 32) {
      v16bf a = load_frag(mb + kc, 256, lane);
      v16bf b = load_frag(WvT + (size_t)nt * 16 * 256 + kc, 256, lane);
      acc = wmma_bf16(a, b, acc);
    }
    int n = nt * 16 + (lane & 15);
#pragma unroll
    for (int g = 0; g < 8; ++g) gatel[(g + 8 * half) * 64 + n] = acc[g];
  } else {
    int w2 = wave - 4;
    for (int t = w2; t < 12; t += 4) {
      int mt = t >> 2, nt = t & 3;
      v8f acc = {};
      for (int kc = 0; kc < 64; kc += 32) {
        v16bf a = load_frag(va + mt * 16 * 64 + kc, 64, lane);
        v16bf b = load_frag(WvvT + (size_t)nt * 16 * 64 + kc, 64, lane);
        acc = wmma_bf16(a, b, acc);
      }
      int n = nt * 16 + (lane & 15);
#pragma unroll
      for (int g = 0; g < 8; ++g)
        hv[(mt * 16 + g + 8 * half) * 64 + n] = acc[g];
    }
  }
  __syncthreads();

  // scatter m into agg_sum[dst]
  for (int i = tid; i < 16 * 256; i += 256) {
    int e = i >> 8, c = i & 255;
    atomicAdd(&agg_sum[(size_t)dstl[e] * SDIM + c], (float)mb[i]);
  }
  // scatter vmsg = rn * gate + hv into v_sum[dst]
  for (int i = tid; i < 48 * 64; i += 256) {
    int r = i >> 6, k = i & 63;
    int e = r / 3, dd = r - e * 3;
    float vm = rnl[r] * gatel[e * 64 + k] + hv[i];
    atomicAdd(&v_sum[((size_t)dstl[e] * 3 + dd) * VDIM + k], vm);
  }
}

// ---------------------------------------------------------------- node update + LN
__global__ void layer_node(const float* __restrict__ agg_sum, const float* __restrict__ deg,
                           const __bf16* __restrict__ WsT, const float* __restrict__ bs,
                           const float* __restrict__ lng, const float* __restrict__ lnb,
                           float* __restrict__ s, __bf16* __restrict__ s_bf) {
  __shared__ __bf16 u[16 * 256];
  __shared__ float  sn[16 * 256];
  __shared__ float  mu[16], rsig[16];
  int m0 = blockIdx.x * 16;
  int tid = threadIdx.x, lane = tid & 31, wave = tid >> 5;
  int half = (lane >> 4) & 1;

  for (int i = tid; i < 16 * 256; i += 256) {
    int r = i >> 8, c = i & 255;
    float dv = fmaxf(deg[m0 + r], 1.0f);
    u[i] = (__bf16)(agg_sum[(size_t)(m0 + r) * SDIM + c] / dv);
  }
  __syncthreads();

  for (int nt = wave; nt < 16; nt += 8) {
    v8f acc = {};
    for (int kc = 0; kc < 256; kc += 32) {
      v16bf a = load_frag(u + kc, 256, lane);
      v16bf b = load_frag(WsT + (size_t)nt * 16 * 256 + kc, 256, lane);
      acc = wmma_bf16(a, b, acc);
    }
    int n = nt * 16 + (lane & 15);
    float bb = bs[n];
#pragma unroll
    for (int g = 0; g < 8; ++g) {
      int r = g + 8 * half;
      sn[r * 256 + n] = s[(size_t)(m0 + r) * SDIM + n] + siluf(acc[g] + bb);
    }
  }
  __syncthreads();

  if (tid < 16) {
    float m = 0.0f;
    for (int c = 0; c < 256; ++c) m += sn[tid * 256 + c];
    m *= (1.0f / 256.0f);
    float v = 0.0f;
    for (int c = 0; c < 256; ++c) { float d = sn[tid * 256 + c] - m; v += d * d; }
    v *= (1.0f / 256.0f);
    mu[tid] = m;
    rsig[tid] = rsqrtf(v + 1e-5f);
  }
  __syncthreads();

  for (int i = tid; i < 16 * 256; i += 256) {
    int r = i >> 8, c = i & 255;
    float val = (sn[i] - mu[r]) * rsig[r] * lng[c] + lnb[c];
    size_t o = (size_t)(m0 + r) * SDIM + c;
    s[o] = val;
    s_bf[o] = (__bf16)val;
  }
}

__global__ void v_update(float* __restrict__ v, const float* __restrict__ v_sum,
                         const float* __restrict__ deg, __bf16* __restrict__ v_bf) {
  int i = blockIdx.x * 256 + threadIdx.x;
  if (i >= N_NODES * 3 * VDIM) return;
  int n = i / (3 * VDIM);
  float dv = fmaxf(deg[n], 1.0f);
  float nv = v[i] + v_sum[i] / dv;
  v[i] = nv;
  v_bf[i] = (__bf16)nv;
}

// ---------------------------------------------------------------- head
__global__ void coords1(const float* __restrict__ v, const float* __restrict__ coW,
                        float* __restrict__ cv) {
  int idx = blockIdx.x * 256 + threadIdx.x;
  if (idx >= N_NODES * 3) return;
  float a = 0.0f;
#pragma unroll
  for (int k = 0; k < VDIM; ++k) a += v[(size_t)idx * VDIM + k] * coW[k];
  cv[idx] = a;
}

__global__ void coords_final(const float* __restrict__ cv, const float* __restrict__ gsum2,
                             const int* __restrict__ batch, const float* __restrict__ pos_c,
                             float* __restrict__ coords_ws, float* __restrict__ out) {
  int n = blockIdx.x * 256 + threadIdx.x;
  if (n >= N_NODES) return;
  int g = batch[n];
  float c = fmaxf(gsum2[g * 4 + 3], 1.0f);
#pragma unroll
  for (int d = 0; d < 3; ++d) {
    float val = pos_c[n * 3 + d] + cv[n * 3 + d] - gsum2[g * 4 + d] / c;
    coords_ws[n * 3 + d] = val;
    out[n * 3 + d] = val;
  }
}

__global__ void bond_kernel(const int* __restrict__ eig, const float* __restrict__ coords,
                            const float* __restrict__ s2,
                            const __bf16* __restrict__ b0WT, const float* __restrict__ b0b,
                            const float* __restrict__ b1W, const float* __restrict__ b1b,
                            float* __restrict__ bonds_out) {
  __shared__ __bf16 feat[16 * 288];
  __shared__ float  h[16 * 256];
  __shared__ int    ii[16], jj[16];
  __shared__ float  dl[16];
  int e0 = blockIdx.x * 16;
  int tid = threadIdx.x, lane = tid & 31, wave = tid >> 5;
  int half = (lane >> 4) & 1;

  if (tid < 16) {
    int e = e0 + tid;
    int j = eig[e], i2 = eig[E_GLOBAL + e];
    ii[tid] = i2; jj[tid] = j;
    float dx = coords[i2 * 3 + 0] - coords[j * 3 + 0];
    float dy = coords[i2 * 3 + 1] - coords[j * 3 + 1];
    float dz = coords[i2 * 3 + 2] - coords[j * 3 + 2];
    dl[tid] = sqrtf(fmaxf(dx * dx + dy * dy + dz * dz, 1e-12f));
  }
  __syncthreads();

  for (int idx = tid; idx < 16 * 288; idx += 256) {
    int e = idx / 288, k = idx - e * 288;
    float val;
    if (k < 256)       val = s2[(size_t)ii[e] * SDIM + k] + s2[(size_t)jj[e] * SDIM + k];
    else if (k == 256) val = dl[e];
    else               val = 0.0f;
    feat[idx] = (__bf16)val;
  }
  __syncthreads();

  for (int nt = wave; nt < 16; nt += 8) {
    v8f acc = {};
    for (int kc = 0; kc < 288; kc += 32) {
      v16bf a = load_frag(feat + kc, 288, lane);
      v16bf b = load_frag(b0WT + (size_t)nt * 16 * 288 + kc, 288, lane);
      acc = wmma_bf16(a, b, acc);
    }
    int n = nt * 16 + (lane & 15);
    float bb = b0b[n];
#pragma unroll
    for (int g = 0; g < 8; ++g)
      h[(g + 8 * half) * 256 + n] = siluf(acc[g] + bb);
  }
  __syncthreads();

  for (int idx = tid; idx < 16 * 5; idx += 256) {
    int e = idx / 5, o = idx - e * 5;
    float a = b1b[o];
    for (int c = 0; c < 256; ++c) a += h[e * 256 + c] * b1W[c * 5 + o];
    bonds_out[(size_t)(e0 + e) * 5 + o] = a;
  }
}

// ================================================================ host
static inline size_t align_up(size_t x) { return (x + 255) & ~(size_t)255; }

extern "C" void kernel_launch(void* const* d_in, const int* in_sizes, int n_in,
                              void* d_out, int out_size, void* d_ws, size_t ws_size,
                              hipStream_t stream) {
  (void)in_sizes; (void)n_in; (void)out_size; (void)ws_size;
  const float* x       = (const float*)d_in[0];
  const float* t       = (const float*)d_in[1];
  const float* pos     = (const float*)d_in[2];
  const int*   eil     = (const int*)d_in[3];
  const int*   eig     = (const int*)d_in[4];
  const int*   batch   = (const int*)d_in[5];
  const float* time_W  = (const float*)d_in[6];
  const float* time_b  = (const float*)d_in[7];
  const float* atom_W  = (const float*)d_in[8];
  const float* atom_b  = (const float*)d_in[9];
  const float* at_W    = (const float*)d_in[10];
  const float* at_b    = (const float*)d_in[11];
  const float* We      = (const float*)d_in[12];
  const float* be      = (const float*)d_in[13];
  const float* Ws      = (const float*)d_in[14];
  const float* bs      = (const float*)d_in[15];
  const float* Wv      = (const float*)d_in[16];
  const float* Wvv     = (const float*)d_in[17];
  const float* ln_g    = (const float*)d_in[18];
  const float* ln_b    = (const float*)d_in[19];
  const float* sh_W    = (const float*)d_in[20];
  const float* sh_b    = (const float*)d_in[21];
  const float* b0_W    = (const float*)d_in[22];
  const float* b0_b    = (const float*)d_in[23];
  const float* b1_W    = (const float*)d_in[24];
  const float* b1_b    = (const float*)d_in[25];
  const float* co_W    = (const float*)d_in[26];
  const float* at2_W   = (const float*)d_in[27];
  const float* at2_b   = (const float*)d_in[28];
  float* out = (float*)d_out;

  char* W = (char*)d_ws;
  size_t off = 0;
  auto take = [&](size_t bytes) { size_t o = off; off = align_up(off + bytes); return o; };

  float*  pos_c   = (float*)(W + take((size_t)N_NODES * 3 * 4));
  float*  s       = (float*)(W + take((size_t)N_NODES * SDIM * 4));
  __bf16* s_bf    = (__bf16*)(W + take((size_t)N_NODES * SDIM * 2));
  __bf16* tmp_bf  = (__bf16*)(W + take((size_t)N_NODES * SDIM * 2));
  float*  v       = (float*)(W + take((size_t)N_NODES * 3 * VDIM * 4));
  __bf16* v_bf    = (__bf16*)(W + take((size_t)N_NODES * 3 * VDIM * 2));
  float*  v_sum   = (float*)(W + take((size_t)N_NODES * 3 * VDIM * 4));
  float*  agg_sum = (float*)(W + take((size_t)N_NODES * SDIM * 4));
  float*  deg     = (float*)(W + take((size_t)N_NODES * 4));
  float*  gsum    = (float*)(W + take((size_t)N_GRAPHS * 4 * 4));
  float*  gsum2   = (float*)(W + take((size_t)N_GRAPHS * 4 * 4));
  int*    src_all = (int*)(W + take((size_t)E_TOT * 4));
  int*    dst_all = (int*)(W + take((size_t)E_TOT * 4));
  float*  env_all = (float*)(W + take((size_t)E_TOT * 4));
  float*  rn_all  = (float*)(W + take((size_t)E_TOT * 3 * 4));
  __bf16* rbf_bf  = (__bf16*)(W + take((size_t)E_TOT * RBF_DIM * 2));
  __bf16* WeT     = (__bf16*)(W + take((size_t)NUM_LAYERS * SDIM * 544 * 2));
  __bf16* WsT     = (__bf16*)(W + take((size_t)NUM_LAYERS * SDIM * SDIM * 2));
  __bf16* WvT     = (__bf16*)(W + take((size_t)NUM_LAYERS * VDIM * SDIM * 2));
  __bf16* WvvT    = (__bf16*)(W + take((size_t)NUM_LAYERS * VDIM * VDIM * 2));
  __bf16* atWT    = (__bf16*)(W + take((size_t)SDIM * SDIM * 2));
  __bf16* shWT    = (__bf16*)(W + take((size_t)SDIM * SDIM * 2));
  __bf16* at2WT   = (__bf16*)(W + take((size_t)16 * SDIM * 2));
  __bf16* b0WT    = (__bf16*)(W + take((size_t)SDIM * 288 * 2));
  float*  s2      = (float*)(W + take((size_t)N_NODES * SDIM * 4));
  __bf16* s2_bf   = (__bf16*)(W + take((size_t)N_NODES * SDIM * 2));
  float*  cv      = (float*)(W + take((size_t)N_NODES * 3 * 4));
  float*  coords  = (float*)(W + take((size_t)N_NODES * 3 * 4));

  const dim3 B(256);
  auto blk = [](size_t n) { return dim3((unsigned)((n + 255) / 256)); };

  // --- weight prep (bf16, transposed [N][Kpad])
  transpose_to_bf16<<<blk(SDIM * SDIM), B, 0, stream>>>(at_W, atWT, SDIM, SDIM, SDIM);
  transpose_to_bf16<<<blk(SDIM * SDIM), B, 0, stream>>>(sh_W, shWT, SDIM, SDIM, SDIM);
  transpose_to_bf16<<<blk(16 * SDIM), B, 0, stream>>>(at2_W, at2WT, SDIM, 16, SDIM);
  transpose_to_bf16<<<blk(SDIM * 288), B, 0, stream>>>(b0_W, b0WT, SDIM + 1, SDIM, 288);
  for (int l = 0; l < NUM_LAYERS; ++l) {
    transpose_to_bf16<<<blk(SDIM * 544), B, 0, stream>>>(
        We + (size_t)l * 544 * SDIM, WeT + (size_t)l * SDIM * 544, 544, SDIM, 544);
    transpose_to_bf16<<<blk(SDIM * SDIM), B, 0, stream>>>(
        Ws + (size_t)l * SDIM * SDIM, WsT + (size_t)l * SDIM * SDIM, SDIM, SDIM, SDIM);
    transpose_to_bf16<<<blk(VDIM * SDIM), B, 0, stream>>>(
        Wv + (size_t)l * SDIM * VDIM, WvT + (size_t)l * VDIM * SDIM, SDIM, VDIM, SDIM);
    transpose_to_bf16<<<blk(VDIM * VDIM), B, 0, stream>>>(
        Wvv + (size_t)l * VDIM * VDIM, WvvT + (size_t)l * VDIM * VDIM, VDIM, VDIM, VDIM);
  }

  // --- pos centering
  (void)hipMemsetAsync(gsum, 0, N_GRAPHS * 4 * sizeof(float), stream);
  accum_graph3<<<blk(N_NODES), B, 0, stream>>>(pos, batch, gsum);
  center_pos<<<blk(N_NODES), B, 0, stream>>>(pos, batch, gsum, pos_c);

  // --- edge geometry + degrees
  (void)hipMemsetAsync(deg, 0, N_NODES * sizeof(float), stream);
  edge_geom<<<blk(E_TOT), B, 0, stream>>>(eil, eig, pos_c, src_all, dst_all,
                                          env_all, rn_all, rbf_bf, deg);

  // --- embedding
  embed1<<<blk((size_t)N_NODES * SDIM), B, 0, stream>>>(x, t, batch, time_W, time_b,
                                                        atom_W, atom_b, tmp_bf);
  gemm_nodes<<<dim3(N_NODES / 16), B, 0, stream>>>(tmp_bf, atWT, at_b, s, s_bf,
                                                   SDIM, SDIM, 0);

  // --- v = 0
  (void)hipMemsetAsync(v, 0, (size_t)N_NODES * 3 * VDIM * sizeof(float), stream);
  (void)hipMemsetAsync(v_bf, 0, (size_t)N_NODES * 3 * VDIM * sizeof(__bf16), stream);

  // --- message passing layers
  for (int l = 0; l < NUM_LAYERS; ++l) {
    (void)hipMemsetAsync(agg_sum, 0, (size_t)N_NODES * SDIM * sizeof(float), stream);
    (void)hipMemsetAsync(v_sum, 0, (size_t)N_NODES * 3 * VDIM * sizeof(float), stream);
    layer_edge<<<dim3(E_TOT / 16), B, 0, stream>>>(
        src_all, dst_all, rn_all, env_all, rbf_bf, s_bf, v_bf,
        WeT + (size_t)l * SDIM * 544, be + (size_t)l * SDIM,
        WvT + (size_t)l * VDIM * SDIM, WvvT + (size_t)l * VDIM * VDIM,
        agg_sum, v_sum);
    layer_node<<<dim3(N_NODES / 16), B, 0, stream>>>(
        agg_sum, deg, WsT + (size_t)l * SDIM * SDIM, bs + (size_t)l * SDIM,
        ln_g + (size_t)l * SDIM, ln_b + (size_t)l * SDIM, s, s_bf);
    v_update<<<blk((size_t)N_NODES * 3 * VDIM), B, 0, stream>>>(v, v_sum, deg, v_bf);
  }

  // --- head
  gemm_nodes<<<dim3(N_NODES / 16), B, 0, stream>>>(s_bf, shWT, sh_b, s2, s2_bf,
                                                   SDIM, SDIM, 1);
  coords1<<<blk((size_t)N_NODES * 3), B, 0, stream>>>(v, co_W, cv);
  (void)hipMemsetAsync(gsum2, 0, N_GRAPHS * 4 * sizeof(float), stream);
  accum_graph3<<<blk(N_NODES), B, 0, stream>>>(cv, batch, gsum2);
  coords_final<<<blk(N_NODES), B, 0, stream>>>(cv, gsum2, batch, pos_c, coords, out);
  gemm_nodes<<<dim3(N_NODES / 16), B, 0, stream>>>(s2_bf, at2WT, at2_b,
                                                   out + (size_t)N_NODES * 3, nullptr,
                                                   SDIM, 16, 0);
  bond_kernel<<<dim3(E_GLOBAL / 16), B, 0, stream>>>(
      eig, coords, s2, b0WT, b0_b, b1_W, b1_b,
      out + (size_t)N_NODES * 3 + (size_t)N_NODES * 16);
}
